// GMPN_59055800320562
// MI455X (gfx1250) — matrix operand order repaired and checked by reading, not compiled, not run on hardware
//
#include <hip/hip_runtime.h>
#include <cstdint>
#include <cstddef>

// ---------------------------------------------------------------------------
// MPNN on MI455X (gfx1250).
// Dominant cost: 10x (A[8192x8192] @ h[8192x32]).
//  - A converted once to bf16 (0/1 exact) -> 128 MB, resident in 192 MB L2.
//  - Per-depth GEMM uses v_wmma_f32_16x16x32_bf16 (wave32, 16x16 tiles),
//    K split into 4 chunks (2048 waves total) so each SIMD has ~4 resident
//    waves to hide L2 latency; partials reduced deterministically.
//  - h kept f32 (master) + bf16 transposed copy HbT[E][N] so the WMMA
//    B-operand is a contiguous 32B load per lane.
// Everything else (<1 GFLOP/depth) runs as plain VALU kernels.
// ---------------------------------------------------------------------------

#define N_ATOMS    8192
#define M_BONDS    16384
#define EMBED      32
#define HIDDEN     256
#define DEPTH      10
#define NUM_GRAPHS 256
#define KSPLIT     4
#define KCHUNK     (N_ATOMS / KSPLIT)   // 2048

typedef __attribute__((ext_vector_type(16))) __bf16        v16bf;
typedef __attribute__((ext_vector_type(8)))  float         v8f;
typedef __attribute__((ext_vector_type(8)))  unsigned short v8u16;
typedef __attribute__((ext_vector_type(16))) unsigned short v16u16;

static __device__ __forceinline__ unsigned short f32_to_bf16(float f) {
  unsigned int u = __float_as_uint(f);
  unsigned int r = 0x7FFFu + ((u >> 16) & 1u);   // round-to-nearest-even
  return (unsigned short)((u + r) >> 16);
}

// ---------------------------------------------------------------------------
// 1) Adjacency int32 -> bf16 (0x3F80 / 0), plus row degree -> inv_deg, has_nei.
// ---------------------------------------------------------------------------
__global__ void k_prepA(const int* __restrict__ adj,
                        unsigned short* __restrict__ A_bf,
                        float* __restrict__ invdeg,
                        int* __restrict__ hasnei) {
  const int row = blockIdx.x;
  const int* a = adj + (size_t)row * N_ATOMS;
  unsigned short* o = A_bf + (size_t)row * N_ATOMS;
  int cnt = 0;
  for (int c = threadIdx.x; c < N_ATOMS; c += 256) {
    int v = (a[c] != 0) ? 1 : 0;
    cnt += v;
    o[c] = v ? (unsigned short)0x3F80 : (unsigned short)0;
  }
  __shared__ int s[256];
  s[threadIdx.x] = cnt;
  __syncthreads();
  for (int off = 128; off > 0; off >>= 1) {
    if (threadIdx.x < off) s[threadIdx.x] += s[threadIdx.x + off];
    __syncthreads();
  }
  if (threadIdx.x == 0) {
    float deg = (float)s[0];
    invdeg[row] = 1.0f / fmaxf(deg, 1.0f);
    hasnei[row] = (s[0] > 0) ? 1 : 0;
  }
}

// ---------------------------------------------------------------------------
// 2) Atom embedding gather: h (f32, row-major [N,E]) + HbT (bf16, [E,N]).
// ---------------------------------------------------------------------------
__global__ void k_embed(const int* __restrict__ af,
                        const float* __restrict__ tab,
                        float* __restrict__ h,
                        unsigned short* __restrict__ HbT) {
  const int idx = blockIdx.x * 256 + threadIdx.x;  // N*E threads
  const int i = idx >> 5, e = idx & 31;
  float v = tab[af[i] * EMBED + e];
  h[idx] = v;
  HbT[(size_t)e * N_ATOMS + i] = f32_to_bf16(v);
}

// ---------------------------------------------------------------------------
// 3a) Partial GEMM: part[chunk] = A[:, kBase:kBase+KCHUNK] @ h[kBase:...]
//     Block = 256 threads = 8 waves; wave w owns a 16-row x 32-col C tile.
//     grid = (64, KSPLIT). A/B operand packing (contiguous k = 16h + s, same for A and B):
//       A (16x32 bf16): lanes 0-15: row=lane, elems {K0..15};
//                       lanes16-31: row=lane-16, elems {K16..31}.
//       B from HbT[n][k]: one contiguous 16-elem (32B) chunk per lane;
//                       lanes 0-15 -> K k0..k0+15, lanes 16-31 -> +16.
// ---------------------------------------------------------------------------
__global__ void k_neiGEMM(const unsigned short* __restrict__ A_bf,
                          const unsigned short* __restrict__ HbT,
                          float* __restrict__ part) {
  const int lane = threadIdx.x & 31;
  const int wave = threadIdx.x >> 5;
  const int rowBase = (blockIdx.x * 8 + wave) * 16;
  const int kBase = blockIdx.y * KCHUNK;
  const int r = lane & 15;
  const bool hi = lane >= 16;

  const unsigned short* aRow = A_bf + (size_t)(rowBase + r) * N_ATOMS;
  const int aOff = hi ? 16 : 0;
  const unsigned short* b0 = HbT + (size_t)r * N_ATOMS + (hi ? 16 : 0);
  const unsigned short* b1 = HbT + (size_t)(r + 16) * N_ATOMS + (hi ? 16 : 0);

  v8f c0 = {}; v8f c1 = {};
  for (int k0 = kBase; k0 < kBase + KCHUNK; k0 += 32) {
    // WGP-scope prefetch (locality 3): pull the L2-resident A stream into
    // the near cache ~4KB ahead of the loads.
    __builtin_prefetch(aRow + k0 + 2048, 0, 3);
    v8u16 alo = *(const v8u16*)(aRow + k0 + aOff);
    v8u16 ahi = *(const v8u16*)(aRow + k0 + 8 + aOff);
    v16u16 a = __builtin_shufflevector(alo, ahi,
        0, 1, 2, 3, 4, 5, 6, 7, 8, 9, 10, 11, 12, 13, 14, 15);
    v16u16 bb0 = *(const v16u16*)(b0 + k0);
    v16u16 bb1 = *(const v16u16*)(b1 + k0);
    c0 = __builtin_amdgcn_wmma_f32_16x16x32_bf16(
        false, __builtin_bit_cast(v16bf, a),
        false, __builtin_bit_cast(v16bf, bb0),
        (short)0, c0, false, false);
    c1 = __builtin_amdgcn_wmma_f32_16x16x32_bf16(
        false, __builtin_bit_cast(v16bf, a),
        false, __builtin_bit_cast(v16bf, bb1),
        (short)0, c1, false, false);
  }
  // C/D layout: VGPR v -> lanes 0-15: (M=v, N=lane); lanes 16-31: (M=8+v).
  float* p = part + (size_t)blockIdx.y * N_ATOMS * EMBED;
#pragma unroll
  for (int v = 0; v < 8; ++v) {
    const int m = hi ? (8 + v) : v;
    const int row = rowBase + m;
    p[(size_t)row * EMBED + r]      = c0[v];
    p[(size_t)row * EMBED + 16 + r] = c1[v];
  }
}

// ---------------------------------------------------------------------------
// 3b) Deterministic K-split reduction + inv_deg scaling.
// ---------------------------------------------------------------------------
__global__ void k_neiReduce(const float* __restrict__ part,
                            const float* __restrict__ invdeg,
                            float* __restrict__ nei) {
  const int idx = blockIdx.x * 256 + threadIdx.x;  // N*E threads
  const int row = idx >> 5;
  float acc = part[idx];
#pragma unroll
  for (int c = 1; c < KSPLIT; ++c)
    acc += part[(size_t)c * N_ATOMS * EMBED + idx];
  nei[idx] = acc * invdeg[row];
}

// ---------------------------------------------------------------------------
// 4) msg = h @ W_self^T + nei_mean @ W_nei^T + b, zeroed for isolated atoms.
// ---------------------------------------------------------------------------
__global__ void k_msg(const float* __restrict__ h,
                      const float* __restrict__ nei,
                      const int* __restrict__ hasnei,
                      const float* __restrict__ W,
                      const float* __restrict__ b,
                      float* __restrict__ msg, int d) {
  __shared__ float hs[EMBED], ns[EMBED];
  const int i = blockIdx.x;
  const int t = threadIdx.x;
  if (t < EMBED)               hs[t]          = h[(size_t)i * EMBED + t];
  else if (t < 2 * EMBED)      ns[t - EMBED]  = nei[(size_t)i * EMBED + (t - EMBED)];
  __syncthreads();
  const float* w = W + ((size_t)d * HIDDEN + t) * (2 * EMBED);
  float acc = b[d * HIDDEN + t];
#pragma unroll
  for (int e = 0; e < EMBED; ++e) acc = fmaf(hs[e], w[e], acc);
#pragma unroll
  for (int e = 0; e < EMBED; ++e) acc = fmaf(ns[e], w[EMBED + e], acc);
  msg[(size_t)i * HIDDEN + t] = hasnei[i] ? acc : 0.0f;
}

// ---------------------------------------------------------------------------
// 5) GRU cell per atom; updates h (f32) in place + refreshes HbT (bf16).
// ---------------------------------------------------------------------------
__global__ void k_gru(const float* __restrict__ msg,
                      float* __restrict__ h,
                      unsigned short* __restrict__ HbT,
                      const float* __restrict__ Wih,
                      const float* __restrict__ Whh,
                      const float* __restrict__ bih,
                      const float* __restrict__ bhh, int d) {
  __shared__ float ms[HIDDEN], hs[EMBED], Gi[96], Gh[96];
  const int i = blockIdx.x;
  const int t = threadIdx.x;
  ms[t]       = msg[(size_t)i * HIDDEN + t];
  ms[t + 128] = msg[(size_t)i * HIDDEN + t + 128];
  if (t < EMBED) hs[t] = h[(size_t)i * EMBED + t];
  __syncthreads();
  if (t < 96) {
    const float* wi = Wih + ((size_t)d * 96 + t) * HIDDEN;
    float gi = bih[d * 96 + t];
    for (int k = 0; k < HIDDEN; ++k) gi = fmaf(ms[k], wi[k], gi);
    const float* wh = Whh + ((size_t)d * 96 + t) * EMBED;
    float gh = bhh[d * 96 + t];
#pragma unroll
    for (int k = 0; k < EMBED; ++k) gh = fmaf(hs[k], wh[k], gh);
    Gi[t] = gi; Gh[t] = gh;
  }
  __syncthreads();
  if (t < EMBED) {
    float rr = 1.0f / (1.0f + __expf(-(Gi[t] + Gh[t])));
    float zz = 1.0f / (1.0f + __expf(-(Gi[32 + t] + Gh[32 + t])));
    float nn = tanhf(Gi[64 + t] + rr * Gh[64 + t]);
    float hn = (1.0f - zz) * nn + zz * hs[t];
    h[(size_t)i * EMBED + t] = hn;
    HbT[(size_t)t * N_ATOMS + i] = f32_to_bf16(hn);
  }
}

// ---------------------------------------------------------------------------
// 6) Bond embedding gather.
// ---------------------------------------------------------------------------
__global__ void k_bond(const int* __restrict__ bf,
                       const float* __restrict__ tab,
                       float* __restrict__ out) {
  const int idx = blockIdx.x * 256 + threadIdx.x;  // M*E threads
  const int m = idx >> 5, e = idx & 31;
  out[idx] = tab[bf[m] * EMBED + e];
}

// ---------------------------------------------------------------------------
// 7) Deterministic segment-mean pooling (batch_indices is sorted).
// ---------------------------------------------------------------------------
__global__ void k_segstart(const int* __restrict__ batch, int* __restrict__ starts) {
  const int g = threadIdx.x;  // 0..255
  int lo = 0, hi = N_ATOMS;
  while (lo < hi) {
    int mid = (lo + hi) >> 1;
    if (batch[mid] < g) lo = mid + 1; else hi = mid;
  }
  starts[g] = lo;
  if (g == 0) starts[NUM_GRAPHS] = N_ATOMS;
}

__global__ void k_poolmean(const float* __restrict__ h,
                           const int* __restrict__ starts,
                           float* __restrict__ gmean) {
  const int g = blockIdx.x, e = threadIdx.x;  // 32 threads
  const int s = starts[g], t = starts[g + 1];
  float acc = 0.0f;
  for (int i = s; i < t; ++i) acc += h[(size_t)i * EMBED + e];
  const int c = t - s;
  gmean[g * EMBED + e] = (c > 0) ? acc / (float)c : 0.0f;
}

__global__ void k_poolout(const float* __restrict__ gmean,
                          const float* __restrict__ pW,
                          const float* __restrict__ pb,
                          float* __restrict__ out) {
  __shared__ float m[EMBED];
  const int g = blockIdx.x, j = threadIdx.x;  // 256 threads
  if (j < EMBED) m[j] = gmean[g * EMBED + j];
  __syncthreads();
  float acc = pb[j];
#pragma unroll
  for (int e = 0; e < EMBED; ++e) acc = fmaf(m[e], pW[j * EMBED + e], acc);
  out[(size_t)g * HIDDEN + j] = acc;
}

// ---------------------------------------------------------------------------
// Host orchestration (stream-ordered, graph-capture safe).
// ---------------------------------------------------------------------------
extern "C" void kernel_launch(void* const* d_in, const int* in_sizes, int n_in,
                              void* d_out, int out_size, void* d_ws, size_t ws_size,
                              hipStream_t stream) {
  (void)in_sizes; (void)n_in; (void)out_size; (void)ws_size;

  const int*   atom_features = (const int*)d_in[0];
  const int*   bond_features = (const int*)d_in[1];
  const int*   adj           = (const int*)d_in[2];
  const int*   batch         = (const int*)d_in[3];
  const float* atom_tab      = (const float*)d_in[4];
  const float* bond_tab      = (const float*)d_in[5];
  const float* msg_W         = (const float*)d_in[6];
  const float* msg_b         = (const float*)d_in[7];
  const float* gru_Wih       = (const float*)d_in[8];
  const float* gru_Whh       = (const float*)d_in[9];
  const float* gru_bih       = (const float*)d_in[10];
  const float* gru_bhh       = (const float*)d_in[11];
  const float* pool_W        = (const float*)d_in[12];
  const float* pool_b        = (const float*)d_in[13];

  // Workspace carve (256-B aligned). Total ~150 MB; A_bf (128 MB) is the
  // L2-resident bf16 adjacency reused across all 10 depths.
  char* ws = (char*)d_ws;
  size_t off = 0;
  auto carve = [&](size_t bytes) -> char* {
    char* p = ws + off;
    off = (off + bytes + 255) & ~(size_t)255;
    return p;
  };
  unsigned short* A_bf   = (unsigned short*)carve((size_t)N_ATOMS * N_ATOMS * 2);
  float*          h      = (float*)carve((size_t)N_ATOMS * EMBED * 4);
  unsigned short* HbT    = (unsigned short*)carve((size_t)EMBED * N_ATOMS * 2);
  float*          nei    = (float*)carve((size_t)N_ATOMS * EMBED * 4);
  float*          part   = (float*)carve((size_t)KSPLIT * N_ATOMS * EMBED * 4);
  float*          msg    = (float*)carve((size_t)N_ATOMS * HIDDEN * 4);
  float*          invdeg = (float*)carve((size_t)N_ATOMS * 4);
  int*            hasnei = (int*)carve((size_t)N_ATOMS * 4);
  float*          gmean  = (float*)carve((size_t)NUM_GRAPHS * EMBED * 4);
  int*            starts = (int*)carve((size_t)(NUM_GRAPHS + 1) * 4);

  float* out_h     = (float*)d_out;
  float* out_bond  = out_h + (size_t)N_ATOMS * EMBED;
  float* out_graph = out_bond + (size_t)M_BONDS * EMBED;

  k_prepA<<<N_ATOMS, 256, 0, stream>>>(adj, A_bf, invdeg, hasnei);
  k_embed<<<(N_ATOMS * EMBED) / 256, 256, 0, stream>>>(atom_features, atom_tab, h, HbT);

  for (int d = 0; d < DEPTH; ++d) {
    k_neiGEMM<<<dim3(N_ATOMS / 128, KSPLIT), 256, 0, stream>>>(A_bf, HbT, part);
    k_neiReduce<<<(N_ATOMS * EMBED) / 256, 256, 0, stream>>>(part, invdeg, nei);
    k_msg<<<N_ATOMS, HIDDEN, 0, stream>>>(h, nei, hasnei, msg_W, msg_b, msg, d);
    k_gru<<<N_ATOMS, 128, 0, stream>>>(msg, h, HbT, gru_Wih, gru_Whh, gru_bih, gru_bhh, d);
  }

  hipMemcpyAsync(out_h, h, (size_t)N_ATOMS * EMBED * sizeof(float),
                 hipMemcpyDeviceToDevice, stream);
  k_bond<<<(M_BONDS * EMBED) / 256, 256, 0, stream>>>(bond_features, bond_tab, out_bond);
  k_segstart<<<1, 256, 0, stream>>>(batch, starts);
  k_poolmean<<<NUM_GRAPHS, 32, 0, stream>>>(h, starts, gmean);
  k_poolout<<<NUM_GRAPHS, HIDDEN, 0, stream>>>(gmean, pool_W, pool_b, out_graph);
}
